// NodeRemovalNet_35596688949388
// MI455X (gfx1250) — matrix operand order repaired and checked by reading, not compiled, not run on hardware
//
#include <hip/hip_runtime.h>
#include <hip/hip_bf16.h>
#include <math.h>

// ---------------------------------------------------------------------------
// Types for CDNA5 WMMA (wave32)
// ---------------------------------------------------------------------------
typedef __attribute__((ext_vector_type(16))) _Float16 v16h;
typedef __attribute__((ext_vector_type(8)))  _Float16 v8h;
typedef __attribute__((ext_vector_type(8)))  float    v8f;

#define B_GRAPHS 16
#define N0_NODES 4096
#define DEG_E    8
#define E_EDGES  (B_GRAPHS * N0_NODES * DEG_E)
#define CW       64

// ---------------------------------------------------------------------------
// Utility kernels
// ---------------------------------------------------------------------------
__global__ void zero_f32_kernel(float* p, int n) {
    int i = blockIdx.x * blockDim.x + threadIdx.x;
    if (i < n) p[i] = 0.0f;
}

__global__ void f32_to_f16_kernel(const float* __restrict__ s, _Float16* __restrict__ d, int n) {
    int i = blockIdx.x * blockDim.x + threadIdx.x;
    if (i < n) d[i] = (_Float16)s[i];
}

__global__ void init_edges_kernel(const int* __restrict__ src_in, const int* __restrict__ dst_in,
                                  int* __restrict__ src, int* __restrict__ dst,
                                  int* __restrict__ valid, int ne) {
    int e = blockIdx.x * blockDim.x + threadIdx.x;
    if (e >= ne) return;
    src[e] = src_in[e];
    dst[e] = dst_in[e];
    valid[e] = 1;
}

// ---------------------------------------------------------------------------
// Edge aggregation (scatter-add with f32 global atomics)
// ---------------------------------------------------------------------------
__global__ void edge_feat_agg_kernel(const float* __restrict__ h, const int* __restrict__ src,
                                     const int* __restrict__ dst, const int* __restrict__ valid,
                                     float* __restrict__ s, int ne, int d) {
    int e = blockIdx.x * blockDim.x + threadIdx.x;
    if (e >= ne) return;
    if (!valid[e]) return;
    int se = src[e], de = dst[e];
    for (int q = 0; q < d; ++q)
        atomicAdd(&s[(size_t)de * d + q], h[(size_t)se * d + q]);
}

__global__ void edge_count_kernel(const int* __restrict__ dst, const int* __restrict__ valid,
                                  float* __restrict__ cnt, int ne) {
    int e = blockIdx.x * blockDim.x + threadIdx.x;
    if (e >= ne) return;
    if (!valid[e]) return;
    atomicAdd(&cnt[dst[e]], 1.0f);
}

__global__ void gcn_edge_agg_kernel(const float* __restrict__ h, const int* __restrict__ src,
                                    const int* __restrict__ dst, const int* __restrict__ valid,
                                    const float* __restrict__ isq, float* __restrict__ s,
                                    int ne, int d) {
    int e = blockIdx.x * blockDim.x + threadIdx.x;
    if (e >= ne) return;
    if (!valid[e]) return;
    int se = src[e], de = dst[e];
    float nrm = isq[se] * isq[de];
    for (int q = 0; q < d; ++q)
        atomicAdd(&s[(size_t)de * d + q], h[(size_t)se * d + q] * nrm);
}

__global__ void invsqrt_deg_kernel(const float* __restrict__ cnt, float* __restrict__ isq, int nn) {
    int i = blockIdx.x * blockDim.x + threadIdx.x;
    if (i < nn) isq[i] = rsqrtf(1.0f + cnt[i]);   // deg includes self-loop
}

// ---------------------------------------------------------------------------
// SAGE layer 0 (K=2, scalar path is optimal)
// ---------------------------------------------------------------------------
__global__ void sage0_kernel(const float* __restrict__ x, const float* __restrict__ s,
                             const float* __restrict__ cnt,
                             const float* __restrict__ wl, const float* __restrict__ bl,
                             const float* __restrict__ wr, float* __restrict__ out, int nn) {
    int idx = blockIdx.x * blockDim.x + threadIdx.x;
    if (idx >= nn * CW) return;
    int i = idx >> 6, j = idx & 63;
    float c  = fmaxf(cnt[i], 1.0f);
    float m0 = s[i * 2 + 0] / c, m1 = s[i * 2 + 1] / c;
    float x0 = x[i * 2 + 0],     x1 = x[i * 2 + 1];
    float v = m0 * wl[j * 2 + 0] + m1 * wl[j * 2 + 1] + bl[j]
            + x0 * wr[j * 2 + 0] + x1 * wr[j * 2 + 1];
    out[idx] = fmaxf(v, 0.0f);
}

// Build A = [mean | x] in f16 for SAGE layers 1-2 (K becomes 2*d)
__global__ void build_a16_sage_kernel(const float* __restrict__ s, const float* __restrict__ cnt,
                                      const float* __restrict__ h, _Float16* __restrict__ a16,
                                      int nn, int d) {
    int idx = blockIdx.x * blockDim.x + threadIdx.x;
    if (idx >= nn * d) return;
    int i = idx / d, q = idx % d;
    float c = fmaxf(cnt[i], 1.0f);
    a16[(size_t)i * 2 * d + q]     = (_Float16)(s[idx] / c);
    a16[(size_t)i * 2 * d + d + q] = (_Float16)h[idx];
}

// Pack Wcat = [wl | wr] (row-major [dout x 2*din]) in f16
__global__ void pack_sage_w_kernel(const float* __restrict__ wl, const float* __restrict__ wr,
                                   _Float16* __restrict__ w16, int dout, int din) {
    int idx = blockIdx.x * blockDim.x + threadIdx.x;
    if (idx >= dout * din) return;
    int j = idx / din, t = idx % din;
    w16[(size_t)j * 2 * din + t]       = (_Float16)wl[idx];
    w16[(size_t)j * 2 * din + din + t] = (_Float16)wr[idx];
}

// ---------------------------------------------------------------------------
// WMMA GEMM: out[M,N] = act(A[M,K] @ Wt[N,K]^T + bias)
// One wave per 16x16 output tile, K stepped by 32, f16 in / f32 accumulate.
// A / B use contiguous k order (k = 16*hi + i); 32-bit C-D VGPR layout per ISA (CDNA5 ISA).
// Requires: M,N multiples of 16; K multiple of 32; rows 32B-aligned.
// ---------------------------------------------------------------------------
__global__ void gemm_wmma_kernel(const _Float16* __restrict__ A,
                                 const _Float16* __restrict__ Wt,
                                 const float* __restrict__ bias,
                                 float* __restrict__ out,
                                 int M, int N, int K, int relu) {
    int lane = threadIdx.x & 31;
    int lo = lane & 15, hi = lane >> 4;
    int m0 = blockIdx.x * 16;
    int n0 = blockIdx.y * 16;

    const _Float16* Arow = A  + (size_t)(m0 + lo) * K;
    const _Float16* Wrow = Wt + (size_t)(n0 + lo) * K;

    v8f acc = {};
    for (int k0 = 0; k0 < K; k0 += 32) {
        if (k0 + 32 < K) __builtin_prefetch(Arow + k0 + 32, 0, 0);
        // A 16x32 f16: lane = row (lo); hi selects K 0..15 vs 16..31; a[i] = A[lo][hi*16+i]
        v16h a = *(const v16h*)(Arow + k0 + hi * 16);
        // B 32x16 f16: lane = column (lo); hi selects K 0..15 vs 16..31; b[i] = B[hi*16+i][lo]
        v16h b = *(const v16h*)(Wrow + k0 + hi * 16);
        acc = __builtin_amdgcn_wmma_f32_16x16x32_f16(
            /*neg_a=*/false, a, /*neg_b=*/false, b,
            /*c_mod=*/(short)0, acc, /*reuse_a=*/false, /*reuse_b=*/false);
    }
#pragma unroll
    for (int r = 0; r < 8; ++r) {
        int m = m0 + hi * 8 + r;   // D: lanes 0-15 -> M=r, lanes 16-31 -> M=8+r
        int n = n0 + lo;
        float v = acc[r];
        if (bias) v += bias[n];
        if (relu) v = fmaxf(v, 0.0f);
        out[(size_t)m * N + n] = v;
    }
}

// GCN combine (in place): h = relu(agg + h/deg + b), 1/deg = isq^2
__global__ void gcn_combine_kernel(float* __restrict__ hlin, const float* __restrict__ agg,
                                   const float* __restrict__ isq, const float* __restrict__ b,
                                   int nn) {
    int idx = blockIdx.x * blockDim.x + threadIdx.x;
    if (idx >= nn * CW) return;
    int i = idx >> 6, j = idx & 63;
    float iq = isq[i];
    hlin[idx] = fmaxf(agg[idx] + hlin[idx] * iq * iq + b[j], 0.0f);
}

// ---------------------------------------------------------------------------
// TopK pooling: score, rank-based selection (order-invariant downstream)
// ---------------------------------------------------------------------------
__global__ void score_kernel(const float* __restrict__ h, const float* __restrict__ p,
                             float* __restrict__ score, int nn, int d) {
    int i = blockIdx.x * blockDim.x + threadIdx.x;
    if (i >= nn) return;
    float dot = 0.0f, n2 = 0.0f;
    for (int q = 0; q < d; ++q) {
        float pv = p[q];
        dot += h[(size_t)i * d + q] * pv;
        n2  += pv * pv;
    }
    score[i] = dot * rsqrtf(n2);
}

__global__ void topk_kernel(const float* __restrict__ h, const float* __restrict__ score,
                            float* __restrict__ xnew, int* __restrict__ newidx,
                            int n, int k, int d) {
    __shared__ float ssc[4096];
    int g = blockIdx.x;
    const float* sc = score + (size_t)g * n;
    for (int i = threadIdx.x; i < n; i += blockDim.x) ssc[i] = sc[i];
    __syncthreads();
    for (int i = threadIdx.x; i < n; i += blockDim.x) {
        float si = ssc[i];
        int rank = 0;
        for (int j = 0; j < n; ++j) {
            float sj = ssc[j];
            rank += (sj > si) || (sj == si && j < i);
        }
        int gi = g * n + i;
        if (rank < k) {
            int ni = g * k + rank;
            float t = tanhf(si);
            for (int q = 0; q < d; ++q)
                xnew[(size_t)ni * d + q] = h[(size_t)gi * d + q] * t;
            newidx[gi] = ni;
        } else {
            newidx[gi] = -1;
        }
    }
}

__global__ void edge_reindex_kernel(const int* __restrict__ src, const int* __restrict__ dst,
                                    const int* __restrict__ valid, const int* __restrict__ newidx,
                                    int* __restrict__ src2, int* __restrict__ dst2,
                                    int* __restrict__ valid2, int ne) {
    int e = blockIdx.x * blockDim.x + threadIdx.x;
    if (e >= ne) return;
    int v = valid[e];
    int s2 = v ? newidx[src[e]] : -1;
    int d2 = v ? newidx[dst[e]] : -1;
    int v2 = (v && s2 >= 0 && d2 >= 0) ? 1 : 0;
    src2[e]   = v2 ? s2 : 0;
    dst2[e]   = v2 ? d2 : 0;
    valid2[e] = v2;
}

// Readout: z[g, 0:d] += max over nodes; z[g, d:2d] += mean over nodes
__global__ void readout_kernel(const float* __restrict__ x, float* __restrict__ z, int k, int d) {
    int g = blockIdx.x;
    int t = threadIdx.x;
    if (t >= d) return;
    float mx = -INFINITY, sum = 0.0f;
    for (int i = 0; i < k; ++i) {
        float v = x[((size_t)g * k + i) * d + t];
        mx = fmaxf(mx, v);
        sum += v;
    }
    z[(size_t)g * 2 * d + t]     += mx;
    z[(size_t)g * 2 * d + d + t] += sum / (float)k;
}

__global__ void softmax_kernel(const float* __restrict__ zin, float* __restrict__ out,
                               int cols) {
    __shared__ float red[256];
    int r = blockIdx.x, t = threadIdx.x;
    const float* row = zin + (size_t)r * cols;
    float m = -INFINITY;
    for (int c = t; c < cols; c += blockDim.x) m = fmaxf(m, row[c]);
    red[t] = m; __syncthreads();
    for (int s = blockDim.x >> 1; s > 0; s >>= 1) {
        if (t < s) red[t] = fmaxf(red[t], red[t + s]);
        __syncthreads();
    }
    m = red[0]; __syncthreads();
    float sum = 0.0f;
    for (int c = t; c < cols; c += blockDim.x) sum += expf(row[c] - m);
    red[t] = sum; __syncthreads();
    for (int s = blockDim.x >> 1; s > 0; s >>= 1) {
        if (t < s) red[t] += red[t + s];
        __syncthreads();
    }
    sum = red[0]; __syncthreads();
    for (int c = t; c < cols; c += blockDim.x)
        out[(size_t)r * cols + c] = expf(row[c] - m) / sum;
}

// ---------------------------------------------------------------------------
// Host orchestration
// ---------------------------------------------------------------------------
static inline int cdiv(int a, int b) { return (a + b - 1) / b; }

extern "C" void kernel_launch(void* const* d_in, const int* in_sizes, int n_in,
                              void* d_out, int out_size, void* d_ws, size_t ws_size,
                              hipStream_t stream) {
    (void)in_sizes; (void)n_in; (void)out_size; (void)ws_size;

    const float* x    = (const float*)d_in[0];
    const int*   esrc = (const int*)d_in[1];
    const int*   edst = (const int*)d_in[2];
    const float* wl_[3] = {(const float*)d_in[3], (const float*)d_in[6], (const float*)d_in[9]};
    const float* bl_[3] = {(const float*)d_in[4], (const float*)d_in[7], (const float*)d_in[10]};
    const float* wr_[3] = {(const float*)d_in[5], (const float*)d_in[8], (const float*)d_in[11]};
    const float* wg_[3] = {(const float*)d_in[12], (const float*)d_in[14], (const float*)d_in[16]};
    const float* bg_[3] = {(const float*)d_in[13], (const float*)d_in[15], (const float*)d_in[17]};
    const float* pool_[6] = {(const float*)d_in[18], (const float*)d_in[19], (const float*)d_in[20],
                             (const float*)d_in[21], (const float*)d_in[22], (const float*)d_in[23]};
    const float* lw1 = (const float*)d_in[24]; const float* lb1 = (const float*)d_in[25];
    const float* lw2 = (const float*)d_in[26]; const float* lb2 = (const float*)d_in[27];
    const float* lw3 = (const float*)d_in[28]; const float* lb3 = (const float*)d_in[29];

    // ---- workspace carve (≈53 MB total) ----
    size_t off = 0;
    char* base = (char*)d_ws;
    auto carve = [&](size_t bytes) -> void* {
        bytes = (bytes + 255) & ~(size_t)255;
        void* p = base + off; off += bytes; return p;
    };
    float* hA   = (float*)carve((size_t)65536 * 64 * 4);   // GEMM / conv output
    float* hB   = (float*)carve((size_t)32768 * 64 * 4);   // pooled features (layer input)
    float* aggS = (float*)carve((size_t)32768 * 64 * 4);   // segment-sum scratch
    float* cnt  = (float*)carve((size_t)65536 * 4);
    float* isq  = (float*)carve((size_t)65536 * 4);
    float* scr  = (float*)carve((size_t)65536 * 4);
    int*   nidx = (int*)  carve((size_t)65536 * 4);
    int* srcA = (int*)carve((size_t)E_EDGES * 4);
    int* dstA = (int*)carve((size_t)E_EDGES * 4);
    int* valA = (int*)carve((size_t)E_EDGES * 4);
    int* srcB = (int*)carve((size_t)E_EDGES * 4);
    int* dstB = (int*)carve((size_t)E_EDGES * 4);
    int* valB = (int*)carve((size_t)E_EDGES * 4);
    _Float16* A16   = (_Float16*)carve((size_t)32768 * 128 * 2);
    _Float16* W16   = (_Float16*)carve((size_t)64 * 128 * 2);
    _Float16* lw1h  = (_Float16*)carve((size_t)128 * 128 * 2);
    _Float16* lw2h  = (_Float16*)carve((size_t)64 * 128 * 2);
    _Float16* lw3h  = (_Float16*)carve((size_t)256 * 64 * 2);
    float* z  = (float*)carve((size_t)16 * 128 * 4);
    float* z1 = (float*)carve((size_t)16 * 128 * 4);
    float* z2 = (float*)carve((size_t)16 * 64 * 4);
    float* z3 = (float*)carve((size_t)16 * 256 * 4);
    _Float16* zh  = (_Float16*)carve((size_t)16 * 128 * 2);
    _Float16* z1h = (_Float16*)carve((size_t)16 * 128 * 2);
    _Float16* z2h = (_Float16*)carve((size_t)16 * 64 * 2);

    const int T = 256;
    auto zero = [&](float* p, int n) {
        zero_f32_kernel<<<cdiv(n, T), T, 0, stream>>>(p, n);
    };

    // edges: copy inputs into mutable ping-pong buffers
    init_edges_kernel<<<cdiv(E_EDGES, T), T, 0, stream>>>(esrc, edst, srcA, dstA, valA, E_EDGES);
    int *sc_ = srcA, *dc_ = dstA, *vc_ = valA;
    int *sn_ = srcB, *dn_ = dstB, *vn_ = valB;

    zero(z, 16 * 128);

    int n = N0_NODES;

    // ---------------- Layer 0: SAGE (din=2) ----------------
    {
        int nn = B_GRAPHS * n;
        zero(aggS, nn * 2);
        zero(cnt, nn);
        edge_feat_agg_kernel<<<cdiv(E_EDGES, T), T, 0, stream>>>(x, sc_, dc_, vc_, aggS, E_EDGES, 2);
        edge_count_kernel<<<cdiv(E_EDGES, T), T, 0, stream>>>(dc_, vc_, cnt, E_EDGES);
        sage0_kernel<<<cdiv(nn * 64, T), T, 0, stream>>>(x, aggS, cnt, wl_[0], bl_[0], wr_[0], hA, nn);
        int k = n / 2;
        score_kernel<<<cdiv(nn, T), T, 0, stream>>>(hA, pool_[0], scr, nn, 64);
        topk_kernel<<<B_GRAPHS, T, 0, stream>>>(hA, scr, hB, nidx, n, k, 64);
        edge_reindex_kernel<<<cdiv(E_EDGES, T), T, 0, stream>>>(sc_, dc_, vc_, nidx, sn_, dn_, vn_, E_EDGES);
        { int* t0=sc_; sc_=sn_; sn_=t0; t0=dc_; dc_=dn_; dn_=t0; t0=vc_; vc_=vn_; vn_=t0; }
        readout_kernel<<<B_GRAPHS, 64, 0, stream>>>(hB, z, k, 64);
        n = k;
    }

    // ---------------- Layers 1-2: SAGE (din=64) via WMMA ----------------
    for (int L = 1; L <= 2; ++L) {
        int nn = B_GRAPHS * n;
        zero(aggS, nn * 64);
        zero(cnt, nn);
        edge_feat_agg_kernel<<<cdiv(E_EDGES, T), T, 0, stream>>>(hB, sc_, dc_, vc_, aggS, E_EDGES, 64);
        edge_count_kernel<<<cdiv(E_EDGES, T), T, 0, stream>>>(dc_, vc_, cnt, E_EDGES);
        build_a16_sage_kernel<<<cdiv(nn * 64, T), T, 0, stream>>>(aggS, cnt, hB, A16, nn, 64);
        pack_sage_w_kernel<<<cdiv(64 * 64, T), T, 0, stream>>>(wl_[L], wr_[L], W16, 64, 64);
        gemm_wmma_kernel<<<dim3(nn / 16, 64 / 16), 32, 0, stream>>>(A16, W16, bl_[L], hA, nn, 64, 128, 1);
        int k = n / 2;
        score_kernel<<<cdiv(nn, T), T, 0, stream>>>(hA, pool_[L], scr, nn, 64);
        topk_kernel<<<B_GRAPHS, T, 0, stream>>>(hA, scr, hB, nidx, n, k, 64);
        edge_reindex_kernel<<<cdiv(E_EDGES, T), T, 0, stream>>>(sc_, dc_, vc_, nidx, sn_, dn_, vn_, E_EDGES);
        { int* t0=sc_; sc_=sn_; sn_=t0; t0=dc_; dc_=dn_; dn_=t0; t0=vc_; vc_=vn_; vn_=t0; }
        readout_kernel<<<B_GRAPHS, 64, 0, stream>>>(hB, z, k, 64);
        n = k;
    }

    // ---------------- Layers 3-5: GCN via WMMA ----------------
    for (int L = 0; L < 3; ++L) {
        int nn = B_GRAPHS * n;
        f32_to_f16_kernel<<<cdiv(nn * 64, T), T, 0, stream>>>(hB, A16, nn * 64);
        f32_to_f16_kernel<<<cdiv(64 * 64, T), T, 0, stream>>>(wg_[L], W16, 64 * 64);
        gemm_wmma_kernel<<<dim3(nn / 16, 64 / 16), 32, 0, stream>>>(A16, W16, nullptr, hA, nn, 64, 64, 0);
        zero(cnt, nn);
        edge_count_kernel<<<cdiv(E_EDGES, T), T, 0, stream>>>(dc_, vc_, cnt, E_EDGES);
        invsqrt_deg_kernel<<<cdiv(nn, T), T, 0, stream>>>(cnt, isq, nn);
        zero(aggS, nn * 64);
        gcn_edge_agg_kernel<<<cdiv(E_EDGES, T), T, 0, stream>>>(hA, sc_, dc_, vc_, isq, aggS, E_EDGES, 64);
        gcn_combine_kernel<<<cdiv(nn * 64, T), T, 0, stream>>>(hA, aggS, isq, bg_[L], nn);
        int k = n / 2;
        score_kernel<<<cdiv(nn, T), T, 0, stream>>>(hA, pool_[3 + L], scr, nn, 64);
        topk_kernel<<<B_GRAPHS, T, 0, stream>>>(hA, scr, hB, nidx, n, k, 64);
        edge_reindex_kernel<<<cdiv(E_EDGES, T), T, 0, stream>>>(sc_, dc_, vc_, nidx, sn_, dn_, vn_, E_EDGES);
        { int* t0=sc_; sc_=sn_; sn_=t0; t0=dc_; dc_=dn_; dn_=t0; t0=vc_; vc_=vn_; vn_=t0; }
        readout_kernel<<<B_GRAPHS, 64, 0, stream>>>(hB, z, k, 64);
        n = k;
    }

    // ---------------- MLP head (WMMA, M=16) + softmax ----------------
    f32_to_f16_kernel<<<cdiv(16 * 128, T), T, 0, stream>>>(z, zh, 16 * 128);
    f32_to_f16_kernel<<<cdiv(128 * 128, T), T, 0, stream>>>(lw1, lw1h, 128 * 128);
    gemm_wmma_kernel<<<dim3(1, 128 / 16), 32, 0, stream>>>(zh, lw1h, lb1, z1, 16, 128, 128, 1);

    f32_to_f16_kernel<<<cdiv(16 * 128, T), T, 0, stream>>>(z1, z1h, 16 * 128);
    f32_to_f16_kernel<<<cdiv(64 * 128, T), T, 0, stream>>>(lw2, lw2h, 64 * 128);
    gemm_wmma_kernel<<<dim3(1, 64 / 16), 32, 0, stream>>>(z1h, lw2h, lb2, z2, 16, 64, 128, 1);

    f32_to_f16_kernel<<<cdiv(16 * 64, T), T, 0, stream>>>(z2, z2h, 16 * 64);
    f32_to_f16_kernel<<<cdiv(256 * 64, T), T, 0, stream>>>(lw3, lw3h, 256 * 64);
    gemm_wmma_kernel<<<dim3(1, 256 / 16), 32, 0, stream>>>(z2h, lw3h, lb3, z3, 16, 256, 64, 0);

    softmax_kernel<<<16, 256, 0, stream>>>(z3, (float*)d_out, 256);
}
